// StructureMapper_87351044866611
// MI455X (gfx1250) — matrix operand order repaired and checked by reading, not compiled, not run on hardware
//
#include <hip/hip_runtime.h>
#include <hip/hip_bf16.h>
#include <math.h>

// ---------------------------------------------------------------------------
// StructureMapper for MI455X (gfx1250, wave32, WMMA).
// Heavy GEMMs: v_wmma_f32_16x16x32_bf16, double-buffered LDS staged by the
// CDNA5 async global->LDS engine (ASYNCcnt), one barrier per K slab.
// ---------------------------------------------------------------------------

typedef __attribute__((ext_vector_type(16))) __bf16 v16bf;
typedef __attribute__((ext_vector_type(8)))  float  v8f;

__device__ __forceinline__ unsigned short f2bf(float f) {
  union { float f; unsigned u; } v; v.f = f;
  unsigned r = v.u + 0x7FFFu + ((v.u >> 16) & 1u);   // round-to-nearest-even
  return (unsigned short)(r >> 16);
}
__device__ __forceinline__ float bf2f(unsigned short h) {
  union { unsigned u; float f; } v; v.u = ((unsigned)h) << 16;
  return v.f;
}
__device__ __forceinline__ unsigned pack2(float lo, float hi) {
  return (unsigned)f2bf(lo) | ((unsigned)f2bf(hi) << 16);
}
// Low 32 bits of a generic LDS pointer == LDS byte offset (ISA: flat-LDS
// truncates to addr[31:0]); that is what async VDST expects.
__device__ __forceinline__ unsigned as_lds(const void* p) {
  return (unsigned)(unsigned long long)p;
}

union FragAB { v16bf v; uint4 q[2]; };
union FragC  { v8f   v; float f[8]; };

// ---------------------------------------------------------------------------
// Tiled WMMA GEMM: out[M,N] = act(A[M,K] @ W[N,K]^T + bias)
//   Block tile 128x128, 8 waves (4 in M x 2 in N), wave tile 32x64.
//   Double-buffered LDS; bf16 tiles staged with global_load_async_to_lds_b128
//   (fp32 A converted through VGPRs instead). One s_barrier per K slab.
// ---------------------------------------------------------------------------
template<bool A_F32, int ACT, bool OUTF, bool OUTH>
__global__ __launch_bounds__(256) void k_gemm(
    const void* __restrict__ Aptr, const unsigned short* __restrict__ W,
    const float* __restrict__ bias,
    float* __restrict__ outF, unsigned short* __restrict__ outH,
    int N, int K)
{
  __shared__ __align__(16) unsigned short lA[2][128][40];
  __shared__ __align__(16) unsigned short lB[2][128][40];

  const int tid  = threadIdx.x;
  const int lane = tid & 31;
  const int wid  = tid >> 5;
  const int wm   = wid & 3;        // 4 waves along M
  const int wn   = wid >> 2;       // 2 waves along N
  const int m0   = blockIdx.y * 128;
  const int n0   = blockIdx.x * 128;

  const int lr = tid >> 1;         // staging row (0..127)
  const int lc = (tid & 1) * 16;   // staging col (halves)

  const int ml = lane & 15;        // lane row/col within a 16-wide fragment
  const int kh = lane >> 4;        // K half-select per ISA layout

  FragC acc[2][4];
#pragma unroll
  for (int i = 0; i < 2; i++)
#pragma unroll
    for (int j = 0; j < 4; j++)
#pragma unroll
      for (int r = 0; r < 8; r++) acc[i][j].f[r] = 0.f;

  // Stage one [128 x 32] A slab + one [128 x 32] W slab into LDS buffer sel.
  auto stage = [&](int k0, int sel) {
    if (A_F32) {
      // fp32 input: convert to bf16 through VGPRs while staging.
      const float* srcp = (const float*)Aptr + (size_t)(m0 + lr) * K + k0 + lc;
      float4 x0 = ((const float4*)srcp)[0];
      float4 x1 = ((const float4*)srcp)[1];
      float4 x2 = ((const float4*)srcp)[2];
      float4 x3 = ((const float4*)srcp)[3];
      uint4 p0, p1;
      p0.x = pack2(x0.x, x0.y); p0.y = pack2(x0.z, x0.w);
      p0.z = pack2(x1.x, x1.y); p0.w = pack2(x1.z, x1.w);
      p1.x = pack2(x2.x, x2.y); p1.y = pack2(x2.z, x2.w);
      p1.z = pack2(x3.x, x3.y); p1.w = pack2(x3.z, x3.w);
      *(uint4*)&lA[sel][lr][lc]     = p0;
      *(uint4*)&lA[sel][lr][lc + 8] = p1;
    } else {
      // bf16 input: async DMA global -> LDS (no VGPR transit, ASYNCcnt).
      unsigned ldsA  = as_lds(&lA[sel][lr][lc]);
      unsigned voffA = (unsigned)(((m0 + lr) * K + k0 + lc) * 2);
      asm volatile("global_load_async_to_lds_b128 %0, %1, %2 offset:0"
                   :: "v"(ldsA), "v"(voffA), "s"(Aptr) : "memory");
      asm volatile("global_load_async_to_lds_b128 %0, %1, %2 offset:16"
                   :: "v"(ldsA), "v"(voffA), "s"(Aptr) : "memory");
    }
    {
      unsigned ldsB  = as_lds(&lB[sel][lr][lc]);
      unsigned voffB = (unsigned)(((n0 + lr) * K + k0 + lc) * 2);
      asm volatile("global_load_async_to_lds_b128 %0, %1, %2 offset:0"
                   :: "v"(ldsB), "v"(voffB), "s"(W) : "memory");
      asm volatile("global_load_async_to_lds_b128 %0, %1, %2 offset:16"
                   :: "v"(ldsB), "v"(voffB), "s"(W) : "memory");
    }
  };

  // Prologue: fill buffer 0.
  stage(0, 0);
  asm volatile("s_wait_asynccnt 0x0" ::: "memory");
  __syncthreads();

  const int NK = K >> 5;
  for (int it = 0; it < NK; ++it) {
    const int cur = it & 1;
    // Software pipeline: stage slab it+1 while computing slab it. Writes go
    // to buffer cur^1, whose last readers finished before the previous
    // barrier, so a single end-of-iteration barrier is sufficient.
    if (it + 1 < NK) stage((it + 1) << 5, cur ^ 1);

    // ---- fragments per ISA 16-bit A(16x32)/B(32x16) wave32 layouts
    FragAB af[2], bfr[4];
#pragma unroll
    for (int mi = 0; mi < 2; mi++) {
      const int mr = wm * 32 + mi * 16 + ml;
      af[mi].q[0] = *(const uint4*)&lA[cur][mr][kh * 8];       // K = kh*8..+7
      af[mi].q[1] = *(const uint4*)&lA[cur][mr][16 + kh * 8];  // K = 16+kh*8..+7
    }
#pragma unroll
    for (int ni = 0; ni < 4; ni++) {
      const int nr = wn * 64 + ni * 16 + ml;
      bfr[ni].q[0] = *(const uint4*)&lB[cur][nr][kh * 8];      // K = kh*8..+7
      bfr[ni].q[1] = *(const uint4*)&lB[cur][nr][16 + kh * 8]; // K = 16+kh*8..+7
    }
#pragma unroll
    for (int mi = 0; mi < 2; mi++)
#pragma unroll
      for (int ni = 0; ni < 4; ni++)
        acc[mi][ni].v = __builtin_amdgcn_wmma_f32_16x16x32_bf16(
            false, af[mi].v, false, bfr[ni].v,
            (short)0, acc[mi][ni].v, false, false);

    // Next slab resident + visible to all waves before it is consumed.
    asm volatile("s_wait_asynccnt 0x0" ::: "memory");
    __syncthreads();
  }

  // ---- epilogue: bias, activation, fp32/bf16 stores
  const int mrb = (lane >> 4) * 8;   // C/D layout: lanes 16-31 hold rows +8
#pragma unroll
  for (int mi = 0; mi < 2; mi++) {
#pragma unroll
    for (int ni = 0; ni < 4; ni++) {
      const int n = n0 + wn * 64 + ni * 16 + ml;
      const float bv = bias ? bias[n] : 0.f;
#pragma unroll
      for (int r = 0; r < 8; r++) {
        const int m = m0 + wm * 32 + mi * 16 + mrb + r;
        float v = acc[mi][ni].f[r] + bv;
        if (ACT == 1) v = v / (1.f + __expf(-v));         // SiLU
        if (OUTF) outF[(size_t)m * N + n] = v;
        if (OUTH) outH[(size_t)m * N + n] = f2bf(v);
      }
    }
  }
}

// ---------------------------------------------------------------------------
// Small elementwise / reduction kernels
// ---------------------------------------------------------------------------
__global__ void k_f32_to_bf16(const float* __restrict__ in,
                              unsigned short* __restrict__ out, int n) {
  int i = blockIdx.x * 256 + threadIdx.x;
  if (i < n) out[i] = f2bf(in[i]);
}

__global__ void k_transpose_to_bf16(const float* __restrict__ in,
                                    unsigned short* __restrict__ out, int H) {
  int i = blockIdx.x * 256 + threadIdx.x;
  if (i < H * H) { int k = i / H, h = i % H; out[i] = f2bf(in[h * H + k]); }
}

__global__ void k_mul_bf16(const unsigned short* __restrict__ a,
                           const unsigned short* __restrict__ b,
                           unsigned short* __restrict__ o, size_t n) {
  size_t i = (size_t)blockIdx.x * 256 + threadIdx.x;
  if (i < n) o[i] = f2bf(bf2f(a[i]) * bf2f(b[i]));
}

__global__ void k_gelu_bf16(const float* __restrict__ in,
                            unsigned short* __restrict__ out, size_t n) {
  size_t i = (size_t)blockIdx.x * 256 + threadIdx.x;
  if (i < n) {
    float x = in[i];
    float t = tanhf(0.7978845608028654f * (x + 0.044715f * x * x * x));
    out[i] = f2bf(0.5f * x * (1.f + t));
  }
}

// rmsnorm over N=256, fp32 in -> bf16 out; one block per row
__global__ __launch_bounds__(256) void k_rmsnorm256_bf16(
    const float* __restrict__ in, const float* __restrict__ g,
    unsigned short* __restrict__ out) {
  __shared__ float red[256];
  const int t = threadIdx.x;
  const size_t base = (size_t)blockIdx.x * 256;
  float v = in[base + t];
  red[t] = v * v; __syncthreads();
  for (int s = 128; s > 0; s >>= 1) { if (t < s) red[t] += red[t + s]; __syncthreads(); }
  float r = rsqrtf(red[0] / 256.f + 1e-6f);
  out[base + t] = f2bf(v * r * g[t]);
}

// rmsnorm over N=1024, fp32 in-place (final answer in d_out)
__global__ __launch_bounds__(256) void k_rmsnorm1024_inplace(
    float* __restrict__ x, const float* __restrict__ g) {
  __shared__ float red[256];
  const int t = threadIdx.x;
  const size_t base = (size_t)blockIdx.x * 1024;
  float v[4]; float s = 0.f;
#pragma unroll
  for (int i = 0; i < 4; i++) { v[i] = x[base + t * 4 + i]; s += v[i] * v[i]; }
  red[t] = s; __syncthreads();
  for (int st = 128; st > 0; st >>= 1) { if (t < st) red[t] += red[t + st]; __syncthreads(); }
  float r = rsqrtf(red[0] / 1024.f + 1e-6f);
#pragma unroll
  for (int i = 0; i < 4; i++) x[base + t * 4 + i] = v[i] * r * g[t * 4 + i];
}

// ---------------------------------------------------------------------------
// Per-batch kernel: bilinear 6x6 scores, 5 Sinkhorn iters, mapping,
// mapped_src, confidence, and assembly of bf16 "combined" [mapped|tgt].
// One block (256 threads) per batch.
// ---------------------------------------------------------------------------
__global__ __launch_bounds__(256) void k_batch(
    const float* __restrict__ enc,        // [98304,256]: src rows then tgt rows
    const float* __restrict__ tmp,        // [49152,256] = src_enc @ Wbil
    unsigned short* __restrict__ combined,// [49152,512] bf16
    float* __restrict__ outMap,           // [B,36]
    float* __restrict__ outMapped,        // [49152,256]
    float* __restrict__ outConf,          // [B]
    const float* __restrict__ bbil,
    const float* __restrict__ Wc1, const float* __restrict__ bc1)
{
  __shared__ float sSrc[6][256], sTgt[6][256], sTmp[6][256];
  __shared__ float sS[6][6];
  __shared__ float red[256];
  const int b = blockIdx.x, t = threadIdx.x;
  const size_t srcBase = (size_t)b * 6 * 256;
  const size_t tgtBase = ((size_t)49152 + (size_t)b * 6) * 256;

  for (int idx = t; idx < 1536; idx += 256) {
    int i = idx >> 8, h = idx & 255;
    sSrc[i][h] = enc[srcBase + idx];
    sTgt[i][h] = enc[tgtBase + idx];
    sTmp[i][h] = tmp[srcBase + idx];
  }
  __syncthreads();

  if (t < 36) {               // log_alpha = (tmp . tgt^T + bbil) / TEMP
    int i = t / 6, j = t % 6;
    float s = 0.f;
    for (int h = 0; h < 256; h++) s += sTmp[i][h] * sTgt[j][h];
    sS[i][j] = (s + bbil[0]) * 2.0f;   // TEMP = 0.5
  }
  __syncthreads();

  for (int it = 0; it < 5; it++) {     // Sinkhorn: row-lse then col-lse
    if (t < 6) {
      float m = sS[t][0];
      for (int j = 1; j < 6; j++) m = fmaxf(m, sS[t][j]);
      float e = 0.f;
      for (int j = 0; j < 6; j++) e += __expf(sS[t][j] - m);
      float l = m + __logf(e);
      for (int j = 0; j < 6; j++) sS[t][j] -= l;
    }
    __syncthreads();
    if (t < 6) {
      float m = sS[0][t];
      for (int i = 1; i < 6; i++) m = fmaxf(m, sS[i][t]);
      float e = 0.f;
      for (int i = 0; i < 6; i++) e += __expf(sS[i][t] - m);
      float l = m + __logf(e);
      for (int i = 0; i < 6; i++) sS[i][t] -= l;
    }
    __syncthreads();
  }

  if (t < 36) {
    float mp = __expf(sS[t / 6][t % 6]);
    sS[t / 6][t % 6] = mp;
    outMap[(size_t)b * 36 + t] = mp;
  }
  __syncthreads();

  float aTot = 0.f;
  for (int idx = t; idx < 1536; idx += 256) {
    int i = idx >> 8, h = idx & 255;
    float v = 0.f;
#pragma unroll
    for (int j = 0; j < 6; j++) v += sS[i][j] * sTgt[j][h];
    outMapped[srcBase + idx] = v;
    size_t crow = ((size_t)b * 6 + i) * 512;
    combined[crow + h]       = f2bf(v);
    combined[crow + 256 + h] = f2bf(sTgt[i][h]);
    aTot += fabsf(v - sSrc[i][h]);
  }
  red[t] = aTot; __syncthreads();
  for (int s = 128; s > 0; s >>= 1) { if (t < s) red[t] += red[t + s]; __syncthreads(); }
  if (t == 0) {
    float ci = red[0] / 1536.f;          // mean over R*H
    float c  = ci * Wc1[0] + bc1[0];
    outConf[b] = 1.f / (1.f + __expf(-c));
  }
}

// ---------------------------------------------------------------------------
// Host launcher
// ---------------------------------------------------------------------------
extern "C" void kernel_launch(void* const* d_in, const int* in_sizes, int n_in,
                              void* d_out, int out_size, void* d_ws, size_t ws_size,
                              hipStream_t stream) {
  (void)in_sizes; (void)n_in; (void)out_size; (void)ws_size;

  const float* src  = (const float*)d_in[0];
  const float* tgt  = (const float*)d_in[1];
  const float* W1   = (const float*)d_in[2];
  const float* b1   = (const float*)d_in[3];
  const float* g1v  = (const float*)d_in[4];
  const float* Wa   = (const float*)d_in[5];
  const float* Wb   = (const float*)d_in[6];
  const float* Wc   = (const float*)d_in[7];
  const float* W2   = (const float*)d_in[8];
  const float* b2   = (const float*)d_in[9];
  const float* Wbil = (const float*)d_in[10];
  const float* bbil = (const float*)d_in[11];
  const float* Wa1  = (const float*)d_in[12];
  const float* ba1  = (const float*)d_in[13];
  const float* Wa2  = (const float*)d_in[14];
  const float* ba2  = (const float*)d_in[15];
  const float* g2v  = (const float*)d_in[16];
  const float* Wc1  = (const float*)d_in[17];
  const float* bc1  = (const float*)d_in[18];

  constexpr int B = 8192, R = 6, D = 1024, H = 256;
  constexpr int Ms = B * R;       // 49152 rows per encoder input
  constexpr int M2 = 2 * Ms;      // 98304 rows (src + tgt stacked)

  // ---- workspace carve (peak ~356 MB) with lifetime-based reuse
  char* ws = (char*)d_ws;
  size_t off = 0;
  auto carve = [&](size_t bytes) -> char* {
    char* p = ws + off; off += (bytes + 255) & ~(size_t)255; return p;
  };
  unsigned short* wW1  = (unsigned short*)carve((size_t)H * D * 2);
  unsigned short* wWa  = (unsigned short*)carve((size_t)2 * H * H * 2);
  unsigned short* wWb  = (unsigned short*)carve((size_t)2 * H * H * 2);
  unsigned short* wWc  = (unsigned short*)carve((size_t)H * 2 * H * 2);
  unsigned short* wW2  = (unsigned short*)carve((size_t)H * H * 2);
  unsigned short* wWbT = (unsigned short*)carve((size_t)H * H * 2);
  unsigned short* wWa1 = (unsigned short*)carve((size_t)2 * H * 2 * H * 2);
  unsigned short* wWa2 = (unsigned short*)carve((size_t)D * 2 * H * 2);
  float*          bufA = (float*)carve((size_t)M2 * H * 4);          // h1 / enc / a1
  unsigned short* bufB = (unsigned short*)carve((size_t)M2 * H * 2); // h1n / enc_bf / gelu
  char*           bufC = carve((size_t)M2 * 2 * H * 2);              // gate->g bf16, then tmp f32
  char*           bufD = carve((size_t)M2 * 2 * H * 2);              // val bf16 / c2 bf16 / combined bf16

  // ---- d_out regions (answer, mapping, mapped_src, confidence)
  float* outAns    = (float*)d_out;                  // [49152,1024]
  float* outMap    = outAns + (size_t)Ms * D;        // [B,36]
  float* outMapped = outMap + (size_t)B * R * R;     // [49152,256]
  float* outConf   = outMapped + (size_t)Ms * H;     // [B]

  dim3 blk(256);

  // 0) weights -> bf16 (Wbil also transposed so tmp = src_enc @ Wbil fits A@W^T)
  k_f32_to_bf16<<<(H * D + 255) / 256, blk, 0, stream>>>(W1, wW1, H * D);
  k_f32_to_bf16<<<(2 * H * H + 255) / 256, blk, 0, stream>>>(Wa, wWa, 2 * H * H);
  k_f32_to_bf16<<<(2 * H * H + 255) / 256, blk, 0, stream>>>(Wb, wWb, 2 * H * H);
  k_f32_to_bf16<<<(2 * H * H + 255) / 256, blk, 0, stream>>>(Wc, wWc, 2 * H * H);
  k_f32_to_bf16<<<(H * H + 255) / 256, blk, 0, stream>>>(W2, wW2, H * H);
  k_transpose_to_bf16<<<(H * H + 255) / 256, blk, 0, stream>>>(Wbil, wWbT, H);
  k_f32_to_bf16<<<(4 * H * H + 255) / 256, blk, 0, stream>>>(Wa1, wWa1, 4 * H * H);
  k_f32_to_bf16<<<(D * 2 * H + 255) / 256, blk, 0, stream>>>(Wa2, wWa2, D * 2 * H);

  // 1) h1 = x @ W1^T + b1  (fp32 inputs consumed directly, K=1024)
  dim3 gEnc1(H / 128, Ms / 128);
  k_gemm<true, 0, true, false><<<gEnc1, blk, 0, stream>>>(
      src, wW1, b1, bufA, nullptr, H, D);
  k_gemm<true, 0, true, false><<<gEnc1, blk, 0, stream>>>(
      tgt, wW1, b1, bufA + (size_t)Ms * H, nullptr, H, D);

  // 2) rmsnorm(h1, g1) -> bf16
  k_rmsnorm256_bf16<<<M2, blk, 0, stream>>>(bufA, g1v, bufB);

  // 3) gate = silu(h @ Wa^T), val = h @ Wb^T (both bf16)
  dim3 gGV((2 * H) / 128, M2 / 128);
  k_gemm<false, 1, false, true><<<gGV, blk, 0, stream>>>(
      bufB, wWa, nullptr, nullptr, (unsigned short*)bufC, 2 * H, H);
  k_gemm<false, 0, false, true><<<gGV, blk, 0, stream>>>(
      bufB, wWb, nullptr, nullptr, (unsigned short*)bufD, 2 * H, H);

  // 4) g = gate * val (in place over gate)
  size_t ngv = (size_t)M2 * 2 * H;
  k_mul_bf16<<<(unsigned)((ngv + 255) / 256), blk, 0, stream>>>(
      (unsigned short*)bufC, (unsigned short*)bufD, (unsigned short*)bufC, ngv);

  // 5) c2 = g @ Wc^T -> bf16
  dim3 gDown(H / 128, M2 / 128);
  k_gemm<false, 0, false, true><<<gDown, blk, 0, stream>>>(
      (unsigned short*)bufC, wWc, nullptr, nullptr, (unsigned short*)bufD, H, 2 * H);

  // 6) enc = c2 @ W2^T + b2 -> fp32 (bufA) + bf16 (bufB)
  k_gemm<false, 0, true, true><<<gDown, blk, 0, stream>>>(
      (unsigned short*)bufD, wW2, b2, bufA, bufB, H, H);

  // 7) tmp = src_enc @ Wbil -> fp32 (bufC)
  dim3 gBil(H / 128, Ms / 128);
  k_gemm<false, 0, true, false><<<gBil, blk, 0, stream>>>(
      bufB, wWbT, nullptr, (float*)bufC, nullptr, H, H);

  // 8) per-batch scores + Sinkhorn + mapping + mapped_src + confidence
  k_batch<<<B, blk, 0, stream>>>(bufA, (float*)bufC, (unsigned short*)bufD,
                                 outMap, outMapped, outConf, bbil, Wc1, bc1);

  // 9) a1 = combined @ Wa1^T + ba1 -> fp32 (bufA)
  dim3 gA1((2 * H) / 128, Ms / 128);
  k_gemm<false, 0, true, false><<<gA1, blk, 0, stream>>>(
      (unsigned short*)bufD, wWa1, ba1, bufA, nullptr, 2 * H, 2 * H);

  // 10) gelu(a1) -> bf16 (bufB)
  size_t na1 = (size_t)Ms * 2 * H;
  k_gelu_bf16<<<(unsigned)((na1 + 255) / 256), blk, 0, stream>>>(bufA, bufB, na1);

  // 11) answer_pre = gelu @ Wa2^T + ba2 -> fp32 straight into d_out
  dim3 gA2(D / 128, Ms / 128);
  k_gemm<false, 0, true, false><<<gA2, blk, 0, stream>>>(
      bufB, wWa2, ba2, outAns, nullptr, D, 2 * H);

  // 12) rmsnorm(answer, g2) in place
  k_rmsnorm1024_inplace<<<Ms, blk, 0, stream>>>(outAns, g2v);
}
